// _MXFPLinearBase_75350906241107
// MI455X (gfx1250) — 8-bit matrix operand order repaired and checked by reading, compiled, not run on hardware
//
#include <hip/hip_runtime.h>
#include <hip/hip_bf16.h>


#define BLK 32
#define FMT_MAX 1.984375f

typedef __attribute__((ext_vector_type(8))) int v8i;

__device__ __forceinline__ int naf_dropped(int p, int eff) {
  const int sgn      = p >> 31;
  const unsigned ap  = (unsigned)((p ^ sgn) - sgn);
  const int sh       = __clz((int)ap) - 8;
  const unsigned xs  = ap << sh;
  const unsigned xh  = xs >> 1;
  const unsigned s   = xs + xh;
  const unsigned pos = s & ~xh;
  const unsigned neg = xh & ~s;
  const int width    = 32 - __clz((int)(pos | neg));
  int drop = width - eff;
  drop = drop < 0 ? 0 : drop;
  drop = drop > 31 ? 31 : drop;
  const unsigned nk  = (1u << drop) - 1u;
  const int dlow = (int)((pos & nk) - (neg & nk)) >> sh;
  return (dlow ^ sgn) - sgn;
}

__device__ __forceinline__ int drop4(int xw, int ww, int iw, int effb) {
  int c = 0;
#pragma unroll
  for (int j = 0; j < 4; ++j) {
    const int xi = (int)(signed char)(xw >> (8 * j));
    const int wi = (int)(signed char)(ww >> (8 * j));
    const int it = (iw >> (8 * j)) & 0xff;
    c += naf_dropped(xi * wi, effb - it);
  }
  return c;
}

__global__ void quantize_kernel(const float* __restrict__ t,
                                signed char* __restrict__ q,
                                float* __restrict__ sc,
                                float* __restrict__ lsc,
                                unsigned char* __restrict__ intra,
                                int nblk) {
  int gid = blockIdx.x * blockDim.x + threadIdx.x;
  if (gid >= nblk) return;
  const float* p = t + (size_t)gid * BLK;

  float v[BLK];
  float ma = 0.0f;
#pragma unroll
  for (int i = 0; i < BLK; ++i) {
    v[i] = p[i];
    ma   = fmaxf(ma, fabsf(v[i]));
  }
  float scale = fmaxf(ma / FMT_MAX, 1e-30f);

  int ee[BLK];
  int me = -60;
#pragma unroll
  for (int i = 0; i < BLK; ++i) {
    float qf = rintf((v[i] / scale) * 64.0f);
    int   iv = (int)qf;
    q[(size_t)gid * BLK + i] = (signed char)iv;
    int ai = iv < 0 ? -iv : iv;
    int e  = ai > 0 ? (31 - __clz(ai)) - 6 : -60;
    ee[i]  = e;
    me     = me > e ? me : e;
  }
  if (intra) {
#pragma unroll
    for (int i = 0; i < BLK; ++i)
      intra[(size_t)gid * BLK + i] = (unsigned char)(me - ee[i]);
  }
  sc[gid]  = scale;
  lsc[gid] = log2f(scale);
}

__global__ __launch_bounds__(128) void msd_mx_gemm(
    const signed char*  __restrict__ xq,  const float* __restrict__ xsc,
    const float*        __restrict__ xlg, const unsigned char* __restrict__ xit,
    const signed char*  __restrict__ wq,  const float* __restrict__ wsc,
    const float*        __restrict__ wlg, const float* __restrict__ bias,
    float* __restrict__ out, int K, int OUT) {
  const int nb   = K / BLK;
  const int row0 = blockIdx.x * 16;
  const int col0 = blockIdx.y * 16;
  const int tid  = threadIdx.x;
  const int lane = tid & 31;
  const int wave = tid >> 5;

  __shared__ signed char   ls_x[16 * 1024];
  __shared__ signed char   ls_w[16 * 1024];
  __shared__ unsigned char ls_it[16 * 1024];
  __shared__ float ls_xs[16 * 32];
  __shared__ float ls_ws[16 * 32];
  __shared__ float ls_xl[16 * 32];
  __shared__ float ls_wl[16 * 32];
  __shared__ float ls_acc[16 * 16];

  __builtin_prefetch(bias + col0, 0, 0);

  {
    const int n16 = (16 * K) >> 4;
    const int4* gx = (const int4*)(xq  + (size_t)row0 * K);
    const int4* gw = (const int4*)(wq  + (size_t)col0 * K);
    const int4* gi = (const int4*)(xit + (size_t)row0 * K);
    int4* lx = (int4*)ls_x;
    int4* lw = (int4*)ls_w;
    int4* li = (int4*)ls_it;
    for (int i = tid; i < n16; i += 128) { lx[i] = gx[i]; lw[i] = gw[i]; li[i] = gi[i]; }

    const int nf = 16 * nb;
    const float* gxs = xsc + (size_t)row0 * nb;
    const float* gws = wsc + (size_t)col0 * nb;
    const float* gxl = xlg + (size_t)row0 * nb;
    const float* gwl = wlg + (size_t)col0 * nb;
    for (int i = tid; i < nf; i += 128) {
      ls_xs[i] = gxs[i]; ls_ws[i] = gws[i]; ls_xl[i] = gxl[i]; ls_wl[i] = gwl[i];
    }
    for (int i = tid; i < 256; i += 128) ls_acc[i] = 0.0f;
  }
  __syncthreads();

  const int n     = lane & 15;
  const int mbase = (lane >> 4) * 8;
  const int hi    = (lane >> 4) & 1;

  int kk[8];
#pragma unroll
  for (int r = 0; r < 8; ++r) {
    const int m = mbase + r;
    int emax = -1000000;
    for (int b = 0; b < nb; ++b) {
      int ce = (int)floorf(ls_xl[m * nb + b] + ls_wl[n * nb + b]);
      emax   = emax > ce ? emax : ce;
    }
    kk[r] = 8 + (emax > 0 ? emax : 0) - emax;
  }

  float acc[8];
#pragma unroll
  for (int r = 0; r < 8; ++r) acc[r] = 0.0f;

  for (int b = wave; b < nb; b += 4) {
    const int k0 = b * BLK;

    const signed char* arow = &ls_x[(lane & 15) * K];
    const int abase = k0 + hi * 8;
    v8i A;
    A[0] = *(const int*)(arow + abase);
    A[1] = *(const int*)(arow + abase + 4);
    A[2] = *(const int*)(arow + abase + 16);
    A[3] = *(const int*)(arow + abase + 20);
    A[4] = 0; A[5] = 0; A[6] = 0; A[7] = 0;

    const signed char* brow = &ls_w[n * K];
    const int bbase = k0 + hi * 8;
    v8i B;
    B[0] = *(const int*)(brow + bbase);
    B[1] = *(const int*)(brow + bbase + 4);
    B[2] = *(const int*)(brow + bbase + 16);
    B[3] = *(const int*)(brow + bbase + 20);
    B[4] = 0; B[5] = 0; B[6] = 0; B[7] = 0;

    v8i C = {0, 0, 0, 0, 0, 0, 0, 0};
    v8i D = __builtin_amdgcn_wmma_i32_16x16x64_iu8(true, A, true, B, C,
                                                   false, false);

    const float wsb = ls_ws[n * nb + b];
    const float lwb = ls_wl[n * nb + b];
    const int2* wp2 = (const int2*)(brow + k0);

#pragma unroll
    for (int r = 0; r < 8; ++r) {
      const int m    = mbase + r;
      const int ce   = (int)floorf(ls_xl[m * nb + b] + lwb);
      const int effb = kk[r] + ce;
      const int2* xp2 = (const int2*)(&ls_x[m * K + k0]);
      const int2* ip2 = (const int2*)(&ls_it[m * K + k0]);
      int corr = 0;
      for (int s8 = 0; s8 < BLK / 8; ++s8) {
        const int2 xw = xp2[s8];
        const int2 ww = wp2[s8];
        const int2 iw = ip2[s8];
        corr += drop4(xw.x, ww.x, iw.x, effb);
        corr += drop4(xw.y, ww.y, iw.y, effb);
      }
      acc[r] += ls_xs[m * nb + b] * wsb * (float)(D[r] - corr);
    }
  }

#pragma unroll
  for (int r = 0; r < 8; ++r)
    atomicAdd(&ls_acc[(mbase + r) * 16 + n], acc[r]);
  __syncthreads();

  for (int i = tid; i < 256; i += 128) {
    const int m = i >> 4;
    const int c = i & 15;
    out[(size_t)(row0 + m) * OUT + (col0 + c)] =
        ls_acc[i] * (1.0f / 4096.0f) + bias[col0 + c];
  }
}

extern "C" void kernel_launch(void* const* d_in, const int* in_sizes, int n_in,
                              void* d_out, int out_size, void* d_ws, size_t ws_size,
                              hipStream_t stream) {
  (void)n_in; (void)out_size; (void)ws_size;
  const float* x    = (const float*)d_in[0];
  const float* w    = (const float*)d_in[1];
  const float* bias = (const float*)d_in[2];
  float* out        = (float*)d_out;

  const int OUT = in_sizes[2];
  const int K   = in_sizes[1] / OUT;
  const int N   = in_sizes[0] / K;
  const int nb  = K / BLK;

  char*  ws  = (char*)d_ws;
  size_t off = 0;
  auto carve = [&](size_t bytes) -> char* {
    char* p = ws + off;
    off = (off + bytes + 255) & ~(size_t)255;
    return p;
  };
  signed char*   xq  = (signed char*)  carve((size_t)N * K);
  signed char*   wqp = (signed char*)  carve((size_t)OUT * K);
  unsigned char* xit = (unsigned char*)carve((size_t)N * K);
  float*         xsc = (float*)        carve((size_t)N * nb * 4);
  float*         xlg = (float*)        carve((size_t)N * nb * 4);
  float*         wsc = (float*)        carve((size_t)OUT * nb * 4);
  float*         wlg = (float*)        carve((size_t)OUT * nb * 4);

  const int nbx = N * nb;
  const int nbw = OUT * nb;
  quantize_kernel<<<(nbx + 255) / 256, 256, 0, stream>>>(x, xq, xsc, xlg, xit, nbx);
  quantize_kernel<<<(nbw + 255) / 256, 256, 0, stream>>>(w, wqp, wsc, wlg, nullptr, nbw);

  dim3 grid(N / 16, OUT / 16);
  msd_mx_gemm<<<grid, 128, 0, stream>>>(xq, xsc, xlg, xit,
                                        wqp, wsc, wlg, bias, out, K, OUT);
}
